// HeteroGraphSAGE_34892314313591
// MI455X (gfx1250) — matrix operand order repaired and checked by reading, not compiled, not run on hardware
//
#include <hip/hip_runtime.h>

// ---------------------------------------------------------------------------
// HeteroGraphConv (2-layer) for MI455X / gfx1250.
//   _gc(x, src, dst, W, b) = nd * segsum_dst( ns[src] * (x @ W)[src] ) + b
// (projection hoisted before the scatter by linearity; ns folded into the
//  GEMM epilogue). GEMMs use v_wmma_f32_16x16x32_f16; scatters use
//  global_atomic_add_f32 (accumulators fit in the 192MB L2).
// ---------------------------------------------------------------------------

typedef _Float16 v8h  __attribute__((ext_vector_type(8)));
typedef _Float16 v16h __attribute__((ext_vector_type(16)));
typedef float    v8f  __attribute__((ext_vector_type(8)));

__global__ void zero_f32_kernel(float* __restrict__ p, long long n) {
  long long i = (long long)blockIdx.x * blockDim.x + threadIdx.x;
  if (i < n) p[i] = 0.0f;
}

__global__ void degree_kernel(const int* __restrict__ src, const int* __restrict__ dst,
                              float* __restrict__ deg_s, float* __restrict__ deg_d, int E) {
  int e = blockIdx.x * blockDim.x + threadIdx.x;
  if (e >= E) return;
  atomicAdd(deg_s + src[e], 1.0f);
  atomicAdd(deg_d + dst[e], 1.0f);
}

__global__ void rsqrt_kernel(float* __restrict__ p, long long n) {
  long long i = (long long)blockIdx.x * blockDim.x + threadIdx.x;
  if (i < n) { float d = p[i]; p[i] = (d > 0.0f) ? rsqrtf(d) : 0.0f; }
}

// Y[row, :] = rowscale[row] * (A[N,K] @ W[K,128])[row, :]
// One workgroup = 8 waves, each wave owns a 16-row x 128-col output stripe.
template <int K>
__global__ __launch_bounds__(256)
void gemm_rowscale_wmma(const float* __restrict__ A, const float* __restrict__ W,
                        const float* __restrict__ rowscale, float* __restrict__ Y, int N) {
  // W staged transposed + f16 in LDS: Wt[n*K + k]  (K=128 -> 32KB, K=256 -> 64KB)
  __shared__ __align__(16) _Float16 Wt[128 * K];
  for (int idx = threadIdx.x; idx < 128 * K; idx += 256) {
    int n = idx & 127, k = idx >> 7;               // coalesced global read over n
    Wt[n * K + k] = (_Float16)W[k * 128 + n];
  }
  __syncthreads();

  const int wave = threadIdx.x >> 5;
  const int lane = threadIdx.x & 31;
  const int m0 = blockIdx.x * 128 + wave * 16;
  if (m0 >= N) return;                             // N % 16 == 0, so no ragged tiles
  const int lh = lane & 15;
  const int hi = lane >> 4;                        // 0 or 1 (half-wave)

  const float* arow = A + (size_t)(m0 + lh) * K;

  v8f acc[8];
#pragma unroll
  for (int t = 0; t < 8; ++t)
#pragma unroll
    for (int j = 0; j < 8; ++j) acc[t][j] = 0.0f;

#pragma unroll
  for (int kk = 0; kk < K; kk += 32) {
    // A fragment (16x32 f16): lanes 0-15 hold K {kk..kk+7, kk+16..kk+23} of row m0+lh;
    // lanes 16-31 hold K {kk+8..15, kk+24..31} (ISA 16-bit A layout).
    v16h a;
    {
      const float4* p0 = (const float4*)(arow + kk + hi * 8);
      const float4* p1 = (const float4*)(arow + kk + hi * 8 + 16);
      float4 x0 = p0[0], x1 = p0[1], x2 = p1[0], x3 = p1[1];
      a[0]  = (_Float16)x0.x; a[1]  = (_Float16)x0.y; a[2]  = (_Float16)x0.z; a[3]  = (_Float16)x0.w;
      a[4]  = (_Float16)x1.x; a[5]  = (_Float16)x1.y; a[6]  = (_Float16)x1.z; a[7]  = (_Float16)x1.w;
      a[8]  = (_Float16)x2.x; a[9]  = (_Float16)x2.y; a[10] = (_Float16)x2.z; a[11] = (_Float16)x2.w;
      a[12] = (_Float16)x3.x; a[13] = (_Float16)x3.y; a[14] = (_Float16)x3.z; a[15] = (_Float16)x3.w;
    }
#pragma unroll
    for (int t = 0; t < 8; ++t) {
      // B fragment (32x16 f16): column n0+lh; lanes 0-15 -> K {kk..kk+7, kk+16..kk+23},
      // lanes 16-31 -> K {kk+8..15, kk+24..31} (same order as A).
      const _Float16* wp = &Wt[(size_t)(t * 16 + lh) * K + kk + hi * 8];
      v8h b0 = *(const v8h*)wp;
      v8h b1 = *(const v8h*)(wp + 16);
      v16h b = __builtin_shufflevector(b0, b1, 0, 1, 2, 3, 4, 5, 6, 7,
                                               8, 9, 10, 11, 12, 13, 14, 15);
      acc[t] = __builtin_amdgcn_wmma_f32_16x16x32_f16(
          /*neg_a=*/false, a, /*neg_b=*/false, b,
          /*c_mod=*/(short)0, acc[t], /*reuse_a=*/false, /*reuse_b=*/false);
    }
  }

  // C/D layout: VGPR r -> rows m0+r (lanes 0-15) and m0+8+r (lanes 16-31), col = lh + 16t.
#pragma unroll
  for (int r = 0; r < 8; ++r) {
    int row = m0 + hi * 8 + r;
    float s = rowscale[row];
    float* yrow = Y + (size_t)row * 128;
#pragma unroll
    for (int t = 0; t < 8; ++t) yrow[t * 16 + lh] = s * acc[t][r];
  }
}

// acc[dst, f] += Y[src, f]   (one thread per (edge, feature))
__global__ void scatter_kernel(const int* __restrict__ src, const int* __restrict__ dst,
                               const float* __restrict__ Y, float* __restrict__ acc, int E) {
  long long tid = (long long)blockIdx.x * blockDim.x + threadIdx.x;
  int e = (int)(tid >> 7);
  int f = (int)(tid & 127);
  if (e >= E) return;
  atomicAdd(acc + (size_t)dst[e] * 128 + f, Y[(size_t)src[e] * 128 + f]);
}

__global__ void finalize2_kernel(const float* __restrict__ a1, const float* __restrict__ a2,
                                 const float* __restrict__ nd1, const float* __restrict__ nd2,
                                 const float* __restrict__ b1, const float* __restrict__ b2,
                                 float* __restrict__ out, int N, int do_relu) {
  long long tid = (long long)blockIdx.x * blockDim.x + threadIdx.x;
  int i = (int)(tid >> 7), f = (int)(tid & 127);
  if (i >= N) return;
  size_t o = (size_t)i * 128 + f;
  float v = a1[o] * nd1[i] + b1[f] + a2[o] * nd2[i] + b2[f];
  if (do_relu) v = fmaxf(v, 0.0f);
  out[o] = v;
}

__global__ void finalize1_kernel(const float* __restrict__ a1, const float* __restrict__ nd1,
                                 const float* __restrict__ b1, float* __restrict__ out,
                                 int N, int do_relu) {
  long long tid = (long long)blockIdx.x * blockDim.x + threadIdx.x;
  int i = (int)(tid >> 7), f = (int)(tid & 127);
  if (i >= N) return;
  size_t o = (size_t)i * 128 + f;
  float v = a1[o] * nd1[i] + b1[f];
  if (do_relu) v = fmaxf(v, 0.0f);
  out[o] = v;
}

extern "C" void kernel_launch(void* const* d_in, const int* in_sizes, int n_in,
                              void* d_out, int out_size, void* d_ws, size_t ws_size,
                              hipStream_t stream) {
  (void)n_in; (void)out_size; (void)ws_size;

  const float* x_user  = (const float*)d_in[0];
  const float* x_tweet = (const float*)d_in[1];
  const int* f_src  = (const int*)d_in[2];
  const int* f_dst  = (const int*)d_in[3];
  const int* p_src  = (const int*)d_in[4];
  const int* p_dst  = (const int*)d_in[5];
  const int* pb_src = (const int*)d_in[6];
  const int* pb_dst = (const int*)d_in[7];
  const float* W_f1  = (const float*)d_in[8];  const float* b_f1  = (const float*)d_in[9];
  const float* W_p1  = (const float*)d_in[10]; const float* b_p1  = (const float*)d_in[11];
  const float* W_pb1 = (const float*)d_in[12]; const float* b_pb1 = (const float*)d_in[13];
  const float* W_f2  = (const float*)d_in[14]; const float* b_f2  = (const float*)d_in[15];
  const float* W_p2  = (const float*)d_in[16]; const float* b_p2  = (const float*)d_in[17];
  const float* W_pb2 = (const float*)d_in[18]; const float* b_pb2 = (const float*)d_in[19];

  const int NU = in_sizes[0] / 128;   // 100000 (divisible by 16)
  const int NT = in_sizes[1] / 256;   // 200000 (divisible by 16)
  const int E_f = in_sizes[2], E_p = in_sizes[4], E_pb = in_sizes[6];

  // workspace layout (~310 MB, all f32)
  float* ws    = (float*)d_ws;
  float* ns_f  = ws;               // NU  out-norm, follows
  float* nd_f  = ns_f + NU;        // NU  in-norm,  follows
  float* ns_pb = nd_f + NU;        // NT  out-norm, postedby
  float* nd_pb = ns_pb + NT;       // NU  in-norm,  postedby
  float* ns_p  = nd_pb + NU;       // NU  out-norm, posted
  float* nd_p  = ns_p + NU;        // NT  in-norm,  posted
  float* Y     = nd_p + NT;                  // NT*128 projection buffer (reused 6x)
  float* accA  = Y + (size_t)NT * 128;       // NU*128 (follows acc -> h_user in place)
  float* accB  = accA + (size_t)NU * 128;    // NU*128 (postedby acc, both layers)
  float* accC  = accB + (size_t)NU * 128;    // NT*128 (posted acc -> h_tweet in place)

  float* out_user  = (float*)d_out;                    // NU*128
  float* out_tweet = out_user + (size_t)NU * 128;      // NT*128

  auto zero = [&](float* p, long long n) {
    zero_f32_kernel<<<(unsigned)((n + 255) / 256), 256, 0, stream>>>(p, n);
  };
  auto scat = [&](const int* s, const int* d, float* acc, int E) {
    long long nt = (long long)E * 128;
    scatter_kernel<<<(unsigned)((nt + 255) / 256), 256, 0, stream>>>(s, d, Y, acc, E);
  };

  // ---- degrees / norms -----------------------------------------------------
  const long long normN = 4LL * NU + 2LL * NT;
  zero(ws, normN);
  degree_kernel<<<(E_f  + 255) / 256, 256, 0, stream>>>(f_src,  f_dst,  ns_f,  nd_f,  E_f);
  degree_kernel<<<(E_pb + 255) / 256, 256, 0, stream>>>(pb_src, pb_dst, ns_pb, nd_pb, E_pb);
  degree_kernel<<<(E_p  + 255) / 256, 256, 0, stream>>>(p_src,  p_dst,  ns_p,  nd_p,  E_p);
  rsqrt_kernel<<<(unsigned)((normN + 255) / 256), 256, 0, stream>>>(ws, normN);

  const unsigned gU = (unsigned)((NU + 127) / 128);
  const unsigned gT = (unsigned)((NT + 127) / 128);
  const long long nU = (long long)NU * 128, nT = (long long)NT * 128;

  // ---- layer 1 -------------------------------------------------------------
  zero(accA, nU);
  gemm_rowscale_wmma<128><<<gU, 256, 0, stream>>>(x_user, W_f1, ns_f, Y, NU);
  scat(f_src, f_dst, accA, E_f);

  zero(accB, nU);
  gemm_rowscale_wmma<256><<<gT, 256, 0, stream>>>(x_tweet, W_pb1, ns_pb, Y, NT);
  scat(pb_src, pb_dst, accB, E_pb);

  zero(accC, nT);
  gemm_rowscale_wmma<128><<<gU, 256, 0, stream>>>(x_user, W_p1, ns_p, Y, NU);
  scat(p_src, p_dst, accC, E_p);

  // h_user = relu(accA*nd_f + b_f1 + accB*nd_pb + b_pb1)   (in place in accA)
  finalize2_kernel<<<(unsigned)((nU + 255) / 256), 256, 0, stream>>>(
      accA, accB, nd_f, nd_pb, b_f1, b_pb1, accA, NU, 1);
  // h_tweet = relu(accC*nd_p + b_p1)                        (in place in accC)
  finalize1_kernel<<<(unsigned)((nT + 255) / 256), 256, 0, stream>>>(
      accC, nd_p, b_p1, accC, NT, 1);

  // ---- layer 2 (user acc / tweet acc live directly in d_out) --------------
  zero(out_user, nU);
  gemm_rowscale_wmma<128><<<gU, 256, 0, stream>>>(accA, W_f2, ns_f, Y, NU);
  scat(f_src, f_dst, out_user, E_f);

  zero(accB, nU);
  gemm_rowscale_wmma<128><<<gT, 256, 0, stream>>>(accC, W_pb2, ns_pb, Y, NT);
  scat(pb_src, pb_dst, accB, E_pb);

  zero(out_tweet, nT);
  gemm_rowscale_wmma<128><<<gU, 256, 0, stream>>>(accA, W_p2, ns_p, Y, NU);
  scat(p_src, p_dst, out_tweet, E_p);

  finalize2_kernel<<<(unsigned)((nU + 255) / 256), 256, 0, stream>>>(
      out_user, accB, nd_f, nd_pb, b_f2, b_pb2, out_user, NU, 0);
  finalize1_kernel<<<(unsigned)((nT + 255) / 256), 256, 0, stream>>>(
      out_tweet, nd_p, b_p2, out_tweet, NT, 0);
}
